// RecommenderNet_81003083203133
// MI455X (gfx1250) — matrix operand order repaired and checked by reading, not compiled, not run on hardware
//
#include <hip/hip_runtime.h>
#include <cstdint>

typedef __attribute__((ext_vector_type(16))) __bf16       v16bf;
typedef __attribute__((ext_vector_type(8)))  float        v8f;
typedef __attribute__((ext_vector_type(4)))  unsigned int v4u;
typedef __attribute__((ext_vector_type(8)))  unsigned int v8u;
typedef __attribute__((ext_vector_type(4)))  unsigned int u32x4;
typedef __attribute__((ext_vector_type(8)))  int          i32x8;
typedef __attribute__((ext_vector_type(4)))  int          i32x4;

#define NKEYS  100000
#define NPAD   100096   // padded key count (multiple of 64); pad rows zeroed in prep
#define NQ     2048
#define DDIM   64
#define NCHUNK 25
#define CHUNK  4096
#define MT     128      // queries per workgroup (8 waves x 16)
#define NT     64       // keys per inner iteration
#define TSTRIDE 144     // LDS tile row stride in bytes (16B aligned, conflict-free)

// Load a 32-byte bf16 WMMA fragment from two 16B-aligned LDS locations.
static __device__ __forceinline__ v16bf ld_frag(const char* p0, const char* p1) {
    v4u a = *(const v4u*)p0;
    v4u b = *(const v4u*)p1;
    v8u c = __builtin_shufflevector(a, b, 0, 1, 2, 3, 4, 5, 6, 7);
    return __builtin_bit_cast(v16bf, c);
}

// Async 16B global -> LDS copy (CDNA5 ASYNCcnt path). lds_off = raw LDS byte
// offset (low 32 bits of a flat shared pointer per the aperture mapping).
static __device__ __forceinline__ void async_ld16(unsigned lds_off, const void* gaddr) {
    asm volatile("global_load_async_to_lds_b128 %0, %1, off"
                 :: "v"(lds_off), "v"(gaddr) : "memory");
}
static __device__ __forceinline__ void wait_async0() {
    asm volatile("s_wait_asynccnt 0" ::: "memory");
}

// TDM: DMA a 64x64-element (2B/elem) 2D tile from a strided tensor into LDS,
// inserting 16B of LDS padding per 128B row => TSTRIDE=144 layout in hardware.
// Descriptor per CDNA5 ISA D# groups 0/1 (2D => groups 2/3 zero).
// This toolchain exposes the 6-arg builtin: (g0, g1, g2, g3, g4, cpol).
static __device__ __forceinline__ void tdm_load_tile64(
        unsigned lds_off, const void* gaddr,
        unsigned tensor_d0, unsigned tensor_d1, unsigned d0_stride_elems) {
    const unsigned long long ga = (unsigned long long)(uintptr_t)gaddr;
    u32x4 g0;
    g0[0] = 1u;                                    // count=1 (valid user D#)
    g0[1] = lds_off;                               // lds_addr (bytes)
    g0[2] = (unsigned)(ga & 0xFFFFFFFFu);          // global_addr[31:0]
    g0[3] = (unsigned)((ga >> 32) & 0x1FFFFFFu)    // global_addr[56:32]
          | (2u << 30);                            // type = 2 ("image")
    i32x8 g1;
    g1[0] = (int)((1u << 16)        // data_size = 2 bytes
                | (1u << 20)        // pad_enable
                | (4u << 22)        // pad_interval: 32 dwords (128B)
                | (3u << 25));      // pad_amount: 4 dwords (16B)
    g1[1] = (int)((tensor_d0 & 0xFFFFu) << 16);                       // tensor_dim0[15:0]
    g1[2] = (int)(((tensor_d0 >> 16) & 0xFFFFu)                       // tensor_dim0[31:16]
                | ((tensor_d1 & 0xFFFFu) << 16));                     // tensor_dim1[15:0]
    g1[3] = (int)(((tensor_d1 >> 16) & 0xFFFFu)                       // tensor_dim1[31:16]
                | (64u << 16));                                       // tile_dim0 = 64
    g1[4] = 64;                       // tile_dim1 = 64 rows (tile_dim2 = 0)
    g1[5] = (int)d0_stride_elems;     // tensor_dim0_stride[31:0] (elements)
    g1[6] = 0;                        // stride hi / dim1_stride lo
    g1[7] = 0;                        // dim1_stride hi (unused for 2D)
    const i32x4 z4 = {0, 0, 0, 0};
    const i32x8 z8 = {0, 0, 0, 0, 0, 0, 0, 0};
    __builtin_amdgcn_tensor_load_to_lds(g0, g1, z4, z4, z8, 0);
}

// ---- DPP-based reductions across the 16-lane halves of a wave32 ------------
// xor1/2/4 via DPP8 lane selects, xor8 via DPP16 row_ror:8 (== xor 8 mod 16).
#define DPP8_SEL(a,b,c,d,e,f,g,h) \
    ((a)|((b)<<3)|((c)<<6)|((d)<<9)|((e)<<12)|((f)<<15)|((g)<<18)|((h)<<21))

static __device__ __forceinline__ float red_max16(float v) {
    int y;
    y = __builtin_amdgcn_mov_dpp8(__float_as_int(v), DPP8_SEL(1,0,3,2,5,4,7,6));
    v = fmaxf(v, __int_as_float(y));
    y = __builtin_amdgcn_mov_dpp8(__float_as_int(v), DPP8_SEL(2,3,0,1,6,7,4,5));
    v = fmaxf(v, __int_as_float(y));
    y = __builtin_amdgcn_mov_dpp8(__float_as_int(v), DPP8_SEL(4,5,6,7,0,1,2,3));
    v = fmaxf(v, __int_as_float(y));
    y = __builtin_amdgcn_update_dpp(0, __float_as_int(v), 0x128 /*row_ror:8*/, 0xF, 0xF, true);
    v = fmaxf(v, __int_as_float(y));
    return v;
}
static __device__ __forceinline__ float red_sum16(float v) {
    int y;
    y = __builtin_amdgcn_mov_dpp8(__float_as_int(v), DPP8_SEL(1,0,3,2,5,4,7,6));
    v += __int_as_float(y);
    y = __builtin_amdgcn_mov_dpp8(__float_as_int(v), DPP8_SEL(2,3,0,1,6,7,4,5));
    v += __int_as_float(y);
    y = __builtin_amdgcn_mov_dpp8(__float_as_int(v), DPP8_SEL(4,5,6,7,0,1,2,3));
    v += __int_as_float(y);
    y = __builtin_amdgcn_update_dpp(0, __float_as_int(v), 0x128 /*row_ror:8*/, 0xF, 0xF, true);
    v += __int_as_float(y);
    return v;
}

// ---------------------------------------------------------------------------
// S[s][d] = sum over keys of emb[n][d]  (deterministic tree reduction)
// ---------------------------------------------------------------------------
__global__ __launch_bounds__(256)
void sum_keys_kernel(const float* __restrict__ user_emb,
                     const float* __restrict__ item_emb,
                     float* __restrict__ S) {
    const int s = blockIdx.x >> 6;
    const int d = blockIdx.x & 63;
    const float* keys = s ? user_emb : item_emb;  // side0 keys=items, side1 keys=users
    __shared__ float red[256];
    float acc = 0.f;
    for (int n = threadIdx.x; n < NKEYS; n += 256)
        acc += keys[(size_t)n * DDIM + d];
    red[threadIdx.x] = acc;
    __syncthreads();
    for (int st = 128; st > 0; st >>= 1) {
        if (threadIdx.x < (unsigned)st) red[threadIdx.x] += red[threadIdx.x + st];
        __syncthreads();
    }
    if (threadIdx.x == 0) S[s * DDIM + d] = red[0];
}

// ---------------------------------------------------------------------------
// prep: H[s][n][j] = bf16(relu(keys[n]·W[j] + b[j])), KT[s][j][n] = bf16(keys[n][j])
// ---------------------------------------------------------------------------
__global__ __launch_bounds__(256)
void prep_kernel(const float* __restrict__ user_emb, const float* __restrict__ item_emb,
                 const float* __restrict__ uW, const float* __restrict__ ub,
                 const float* __restrict__ iW, const float* __restrict__ ib,
                 __bf16* __restrict__ Hb, __bf16* __restrict__ KTb) {
    const int s = blockIdx.y;
    const float* keys = s ? user_emb : item_emb;
    const float* W    = s ? iW : uW;
    const float* bia  = s ? ib : ub;
    __shared__ float Wsh[DDIM * DDIM];
    __shared__ float bsh[DDIM];
    for (int i = threadIdx.x; i < DDIM * DDIM; i += 256) Wsh[i] = W[i];
    if (threadIdx.x < DDIM) bsh[threadIdx.x] = bia[threadIdx.x];
    __syncthreads();

    const int n = blockIdx.x * 256 + threadIdx.x;
    __bf16* H  = Hb  + (size_t)s * NPAD * DDIM + (size_t)n * DDIM;
    __bf16* KT = KTb + (size_t)s * DDIM * NPAD + n;
    if (n >= NKEYS) {
        const __bf16 z = (__bf16)0.0f;
#pragma unroll
        for (int j = 0; j < DDIM; ++j) { H[j] = z; KT[(size_t)j * NPAD] = z; }
        return;
    }
    float k[DDIM];
    const float* kr = keys + (size_t)n * DDIM;
#pragma unroll
    for (int i = 0; i < DDIM; ++i) k[i] = kr[i];
#pragma unroll
    for (int j = 0; j < DDIM; ++j) KT[(size_t)j * NPAD] = (__bf16)k[j];
    for (int j = 0; j < DDIM; ++j) {
        float acc = bsh[j];
#pragma unroll
        for (int i = 0; i < DDIM; ++i) acc = fmaf(k[i], Wsh[j * DDIM + i], acc);
        H[j] = (__bf16)fmaxf(acc, 0.f);
    }
}

// ---------------------------------------------------------------------------
// gather: Qg[s][q][d] = emb[id[q]][d]
// ---------------------------------------------------------------------------
__global__ __launch_bounds__(256)
void gather_kernel(const float* __restrict__ user_emb, const float* __restrict__ item_emb,
                   const int* __restrict__ user_id, const int* __restrict__ item_id,
                   float* __restrict__ Qg) {
    const int t = blockIdx.x * 256 + threadIdx.x;   // < 2*2048*64 = 262144
    const int s = t >> 17;
    const int r = t & 131071;
    const int q = r >> 6, d = r & 63;
    const int id = s ? item_id[q] : user_id[q];
    const float* e = s ? item_emb : user_emb;
    Qg[t] = e[(size_t)id * DDIM + d];
}

// ---------------------------------------------------------------------------
// Flash-attention split-K kernel (bf16 WMMA; TDM stages H tiles, async-LDS
// stages KT tiles; double buffered). One WG = 128 queries x 4096 keys.
// Writes un-normalized (m, l, O) partials.
// ---------------------------------------------------------------------------
__global__ __launch_bounds__(256)
void attn_kernel(const __bf16* __restrict__ Hb,   // [2][NPAD][64]
                 const __bf16* __restrict__ KTb,  // [2][64][NPAD]
                 const float*  __restrict__ Qg,   // [2][NQ][64]
                 float* __restrict__ pO, float* __restrict__ pm, float* __restrict__ pl) {
    const int s  = blockIdx.z;
    const int mt = blockIdx.y;
    const int c  = blockIdx.x;
    const int tid = threadIdx.x;
    const int w = tid >> 5, lane = tid & 31;
    const int l16 = lane & 15, hi = lane >> 4;

    const __bf16* H  = Hb  + (size_t)s * NPAD * DDIM;
    const __bf16* KT = KTb + (size_t)s * DDIM * NPAD;

    // double-buffered tiles, 144B row stride (16B aligned, conflict-free)
    __shared__ __align__(16) __bf16 Hs[2 * 64 * 72];
    __shared__ __align__(16) __bf16 Ks[2 * 64 * 72];
    __shared__ __align__(16) __bf16 Ps[8 * 16 * 72];

    // --- Q A-fragments (contiguous k order: k = 16*hi + s, same as B), fixed for the whole kernel ---
    v16bf qa[2];
    const int qrow = mt * MT + w * 16 + l16;
    const float* qptr = Qg + ((size_t)s * NQ + qrow) * DDIM;
#pragma unroll
    for (int f = 0; f < 2; ++f) {
#pragma unroll
        for (int i = 0; i < 8; ++i) {
            const int d = f * 32 + 2 * i + (hi ? 16 : 0);
            qa[f][2 * i]     = (__bf16)qptr[d];
            qa[f][2 * i + 1] = (__bf16)qptr[d + 1];
        }
    }

    v8f o[4];
    float m[8], l[8];
#pragma unroll
    for (int i = 0; i < 4; ++i) o[i] = (v8f){0.f, 0.f, 0.f, 0.f, 0.f, 0.f, 0.f, 0.f};
#pragma unroll
    for (int r = 0; r < 8; ++r) { m[r] = -1e30f; l[r] = 0.f; }

    const int cstart = c * CHUNK;
    const int cend   = (cstart + CHUNK < NKEYS) ? (cstart + CHUNK) : NKEYS;

    // this thread's KT staging slice (2 x 16B)
    const int row0 = tid >> 3,          col0 = tid & 7;
    const int row1 = (tid + 256) >> 3,  col1 = tid & 7;

    // stage one 64-key tile pair into buffer `buf`:
    //  - H tile via the Tensor Data Mover (uniform D#; every wave issues the
    //    identical descriptor and waits TENSORcnt, safe under any EXEC lowering)
    //  - KT tile via per-lane async global->LDS b128 copies (ASYNCcnt)
    auto stage = [&](int buf, int kb) {
        tdm_load_tile64((unsigned)(uintptr_t)((char*)Hs + buf * (64 * TSTRIDE)),
                        (const char*)H + (size_t)kb * 128,
                        64u, (unsigned)NPAD, 64u);
        char* kbp = (char*)Ks + buf * (64 * TSTRIDE);
        async_ld16((unsigned)(uintptr_t)(kbp + row0 * TSTRIDE + col0 * 16),
                   (const char*)KT + (size_t)row0 * (NPAD * 2) + (size_t)kb * 2 + col0 * 16);
        async_ld16((unsigned)(uintptr_t)(kbp + row1 * TSTRIDE + col1 * 16),
                   (const char*)KT + (size_t)row1 * (NPAD * 2) + (size_t)kb * 2 + col1 * 16);
    };
    auto wait_stage = [&]() {
        wait_async0();
        __builtin_amdgcn_s_wait_tensorcnt(0);
        __syncthreads();
    };

    // prologue: stage tile 0
    stage(0, cstart);
    wait_stage();
    int cur = 0;

    for (int kb = cstart; kb < cend; kb += NT) {
        // prefetch next tile into the other buffer while we compute
        if (kb + NT < cend) stage(cur ^ 1, kb + NT);

        const char* HsCur = (const char*)Hs + cur * (64 * TSTRIDE);
        const char* KsCur = (const char*)Ks + cur * (64 * TSTRIDE);

        // --- scores P = Q @ hidden^T  (4 key 16-tiles x 2 k-steps) ---
        v8f p[4];
#pragma unroll
        for (int t = 0; t < 4; ++t) {
            v8f acc = (v8f){0.f, 0.f, 0.f, 0.f, 0.f, 0.f, 0.f, 0.f};
#pragma unroll
            for (int f = 0; f < 2; ++f) {
                const char* base = HsCur + (t * 16 + l16) * TSTRIDE + f * 64 + (hi ? 32 : 0);
                v16bf bv = ld_frag(base, base + 16);
                acc = __builtin_amdgcn_wmma_f32_16x16x32_bf16(
                        false, qa[f], false, bv, (short)0, acc, false, false);
            }
            p[t] = acc;
        }
        // mask keys beyond the valid range of this chunk
#pragma unroll
        for (int t = 0; t < 4; ++t) {
            const int key = kb + t * 16 + l16;
            if (key >= cend) {
#pragma unroll
                for (int r = 0; r < 8; ++r) p[t][r] = -1e30f;
            }
        }
        // --- online softmax (rows: M = r + 8*hi; 16 score cols per tile) ---
#pragma unroll
        for (int r = 0; r < 8; ++r) {
            float vmax = fmaxf(fmaxf(p[0][r], p[1][r]), fmaxf(p[2][r], p[3][r]));
            vmax = red_max16(vmax);
            const float mn = fmaxf(m[r], vmax);
            const float alpha = __expf(m[r] - mn);
            m[r] = mn;
            float rs = 0.f;
#pragma unroll
            for (int t = 0; t < 4; ++t) {
                const float e = __expf(p[t][r] - mn);
                p[t][r] = e;
                rs += e;
            }
            rs = red_sum16(rs);
            l[r] = l[r] * alpha + rs;
#pragma unroll
            for (int nt = 0; nt < 4; ++nt) o[nt][r] *= alpha;
        }
        // --- restage exp(P) through per-wave LDS: C-layout -> A-layout ---
        char* pw = (char*)Ps + w * (16 * TSTRIDE);
#pragma unroll
        for (int t = 0; t < 4; ++t) {
#pragma unroll
            for (int r = 0; r < 8; ++r) {
                *(__bf16*)(pw + (r + 8 * hi) * TSTRIDE + (t * 16 + l16) * 2) = (__bf16)p[t][r];
            }
        }
        asm volatile("s_wait_dscnt 0" ::: "memory");  // same-wave cross-lane LDS visibility
        v16bf pa[2];
#pragma unroll
        for (int f = 0; f < 2; ++f) {
            const char* base = pw + l16 * TSTRIDE + f * 64 + (hi ? 32 : 0);
            pa[f] = ld_frag(base, base + 16);
        }
        // --- O += exp(P) @ K  (contraction over 64 keys, B from KT tile) ---
#pragma unroll
        for (int nt = 0; nt < 4; ++nt) {
#pragma unroll
            for (int f = 0; f < 2; ++f) {
                const char* base = KsCur + (nt * 16 + l16) * TSTRIDE + f * 64 + (hi ? 32 : 0);
                v16bf bv = ld_frag(base, base + 16);
                o[nt] = __builtin_amdgcn_wmma_f32_16x16x32_bf16(
                            false, pa[f], false, bv, (short)0, o[nt], false, false);
            }
        }
        // prefetched tile complete + all waves done with `cur`
        wait_stage();
        cur ^= 1;
    }

    // --- write partials ---
    const size_t pbase = (size_t)(s * NCHUNK + c) * NQ;
#pragma unroll
    for (int r = 0; r < 8; ++r) {
        const int q = mt * MT + w * 16 + r + 8 * hi;
#pragma unroll
        for (int nt = 0; nt < 4; ++nt)
            pO[(pbase + q) * DDIM + nt * 16 + l16] = o[nt][r];
    }
    if (l16 == 0) {
#pragma unroll
        for (int r = 0; r < 8; ++r) {
            const int q = mt * MT + w * 16 + r + 8 * hi;
            pm[pbase + q] = m[r];
            pl[pbase + q] = l[r];
        }
    }
}

// ---------------------------------------------------------------------------
// merge partials + epilogue:  q2 = Q + softmax@K ;  out = leakyrelu((q2+S)W1^T + (q2*S)W2^T)
// og[s][q] = [ Q(64) | out(64) ]
// ---------------------------------------------------------------------------
__global__ __launch_bounds__(256)
void merge_epi_kernel(const float* __restrict__ Qg, const float* __restrict__ S,
                      const float* __restrict__ pO, const float* __restrict__ pm,
                      const float* __restrict__ pl,
                      const float* __restrict__ uW1, const float* __restrict__ uW2,
                      const float* __restrict__ iW1, const float* __restrict__ iW2,
                      float* __restrict__ og) {
    const int s  = blockIdx.y;
    const int qi = threadIdx.x >> 6;    // 4 queries per block
    const int d  = threadIdx.x & 63;
    const int q  = blockIdx.x * 4 + qi;
    const float* W1 = s ? iW1 : uW1;
    const float* W2 = s ? iW2 : uW2;
    __shared__ float ush[4][DDIM];
    __shared__ float vsh[4][DDIM];

    const size_t base = (size_t)s * NCHUNK * NQ;
    float mstar = -1e30f;
    for (int c = 0; c < NCHUNK; ++c)
        mstar = fmaxf(mstar, pm[base + (size_t)c * NQ + q]);
    float lstar = 0.f, acc = 0.f;
    for (int c = 0; c < NCHUNK; ++c) {
        const size_t idx = base + (size_t)c * NQ + q;
        const float wgt = __expf(pm[idx] - mstar);
        lstar += wgt * pl[idx];
        acc = fmaf(wgt, pO[idx * DDIM + d], acc);
    }
    const float qv = Qg[((size_t)s * NQ + q) * DDIM + d];
    const float q2 = qv + acc / lstar;
    const float Svd = S[s * DDIM + d];
    ush[qi][d] = q2 + Svd;
    vsh[qi][d] = q2 * Svd;
    __syncthreads();

    float ov = 0.f;
#pragma unroll
    for (int i = 0; i < DDIM; ++i)
        ov += W1[d * DDIM + i] * ush[qi][i] + W2[d * DDIM + i] * vsh[qi][i];
    ov = (ov > 0.f) ? ov : 0.01f * ov;

    float* ogr = og + ((size_t)s * NQ + q) * 128;
    ogr[d]      = qv;
    ogr[64 + d] = ov;
}

// ---------------------------------------------------------------------------
// out[128][128] = og_user^T @ og_item   (exact fp32, LDS-tiled, deterministic)
// ---------------------------------------------------------------------------
__global__ __launch_bounds__(256)
void final_gemm_kernel(const float* __restrict__ og, float* __restrict__ out) {
    const int ti = blockIdx.x >> 3, tj = blockIdx.x & 7;
    const int a = threadIdx.x >> 4, b = threadIdx.x & 15;
    __shared__ float Us[64][16];
    __shared__ float Is[64][16];
    const float* U = og;
    const float* I = og + (size_t)NQ * 128;
    float acc = 0.f;
    for (int q0 = 0; q0 < NQ; q0 += 64) {
        __syncthreads();
#pragma unroll
        for (int r = 0; r < 4; ++r) {
            const int idx = threadIdx.x + r * 256;
            const int row = idx >> 4, col = idx & 15;
            Us[row][col] = U[(size_t)(q0 + row) * 128 + ti * 16 + col];
            Is[row][col] = I[(size_t)(q0 + row) * 128 + tj * 16 + col];
        }
        __syncthreads();
#pragma unroll 8
        for (int qq = 0; qq < 64; ++qq)
            acc = fmaf(Us[qq][a], Is[qq][b], acc);
    }
    out[(size_t)(ti * 16 + a) * 128 + tj * 16 + b] = acc;
}

// ---------------------------------------------------------------------------
extern "C" void kernel_launch(void* const* d_in, const int* in_sizes, int n_in,
                              void* d_out, int out_size, void* d_ws, size_t ws_size,
                              hipStream_t stream) {
    const float* user_emb = (const float*)d_in[0];
    const float* item_emb = (const float*)d_in[1];
    const float* u_attn_W = (const float*)d_in[2];
    const float* u_attn_b = (const float*)d_in[3];
    const float* u_W1     = (const float*)d_in[4];
    const float* u_W2     = (const float*)d_in[5];
    const float* i_attn_W = (const float*)d_in[6];
    const float* i_attn_b = (const float*)d_in[7];
    const float* i_W1     = (const float*)d_in[8];
    const float* i_W2     = (const float*)d_in[9];
    const int*   user_id  = (const int*)d_in[10];
    const int*   item_id  = (const int*)d_in[11];
    (void)in_sizes; (void)n_in; (void)out_size; (void)ws_size;

    size_t off = 0;
    char* ws = (char*)d_ws;
    auto take = [&](size_t bytes) -> char* {
        char* p = ws + off;
        off += (bytes + 255) & ~(size_t)255;
        return p;
    };
    __bf16* Hb  = (__bf16*)take((size_t)2 * NPAD * DDIM * sizeof(__bf16));  // hidden, bf16
    __bf16* KTb = (__bf16*)take((size_t)2 * DDIM * NPAD * sizeof(__bf16));  // keys^T, bf16
    float*  Qg  = (float*)take((size_t)2 * NQ * DDIM * sizeof(float));
    float*  Sv  = (float*)take((size_t)2 * DDIM * sizeof(float));
    float*  og  = (float*)take((size_t)2 * NQ * 128 * sizeof(float));
    float*  pO  = (float*)take((size_t)2 * NCHUNK * NQ * DDIM * sizeof(float));
    float*  pm  = (float*)take((size_t)2 * NCHUNK * NQ * sizeof(float));
    float*  pl  = (float*)take((size_t)2 * NCHUNK * NQ * sizeof(float));

    sum_keys_kernel<<<dim3(128), dim3(256), 0, stream>>>(user_emb, item_emb, Sv);
    prep_kernel<<<dim3(NPAD / 256, 2), dim3(256), 0, stream>>>(
        user_emb, item_emb, u_attn_W, u_attn_b, i_attn_W, i_attn_b, Hb, KTb);
    gather_kernel<<<dim3((2 * NQ * DDIM) / 256), dim3(256), 0, stream>>>(
        user_emb, item_emb, user_id, item_id, Qg);
    attn_kernel<<<dim3(NCHUNK, NQ / MT, 2), dim3(256), 0, stream>>>(
        Hb, KTb, Qg, pO, pm, pl);
    merge_epi_kernel<<<dim3(NQ / 4, 2), dim3(256), 0, stream>>>(
        Qg, Sv, pO, pm, pl, u_W1, u_W2, i_W1, i_W2, og);
    final_gemm_kernel<<<dim3(64), dim3(256), 0, stream>>>(og, (float*)d_out);
}
